// SpaceByteAttention_39015482917180
// MI455X (gfx1250) — matrix operand order repaired and checked by reading, not compiled, not run on hardware
//
#include <hip/hip_runtime.h>
#include <hip/hip_bf16.h>

// ---------------------------------------------------------------------------
// Types for CDNA5 WMMA (wave32): A/B = 16 bf16 per lane, C/D = 8 f32 per lane
// ---------------------------------------------------------------------------
typedef __attribute__((ext_vector_type(16))) __bf16 v16bf;
typedef __attribute__((ext_vector_type(8)))  float  v8f;

union BF16x16 { v16bf v; uint4 q[2]; unsigned short u[16]; };

__device__ __forceinline__ unsigned short f2bf(float f) {
  unsigned u = __float_as_uint(f);
  u += 0x7FFFu + ((u >> 16) & 1u);        // round-to-nearest-even
  return (unsigned short)(u >> 16);
}

// swap halves of the wave (lane ^= 16): ds_swizzle group-of-32, xor=0x10
__device__ __forceinline__ float swap16f(float x) {
  return __int_as_float(__builtin_amdgcn_ds_swizzle(__float_as_int(x), 0x401F));
}

#define WMMA_BF16(A, B, C) \
  __builtin_amdgcn_wmma_f32_16x16x32_bf16(false, (A), false, (B), (short)0, (C), false, false)

// ---------------------------------------------------------------------------
// Shapes
// ---------------------------------------------------------------------------
#define BATCH   2
#define SEQ     2048
#define DIM     768
#define HD      64
#define NH      12
#define NKV     4
#define GROUPS  3
#define WIN     384
#define MROWS   (BATCH * SEQ)          // 4096
#define NQKV    (NH * HD + 2 * NKV * HD)  // 1280

// ---------------------------------------------------------------------------
// fp32 -> bf16 elementwise convert
// ---------------------------------------------------------------------------
__global__ __launch_bounds__(256) void cvt_f32_bf16(const float* __restrict__ src,
                                                    unsigned short* __restrict__ dst,
                                                    int n) {
  int i = blockIdx.x * 256 + threadIdx.x;
  if (i < n) dst[i] = f2bf(src[i]);
}

// ---------------------------------------------------------------------------
// GEMM: out[m][n] = sum_k X[m][k] * W[n][k]   (X: MxK bf16, W: NxK bf16, out f32)
// Workgroup = 4 waves; wave computes a 16x64 tile with 4x v_wmma_f32_16x16x32_bf16
// per K-step. A operand: lane = row, two 16B chunks at k+8*hi and k+16+8*hi.
// B operand: lane = col, two 16B chunks of the weight row at k+8*hi and k+16+8*hi.
// ---------------------------------------------------------------------------
__global__ __launch_bounds__(128) void gemm_xwt_bf16(const unsigned short* __restrict__ X,
                                                     const unsigned short* __restrict__ W,
                                                     float* __restrict__ out,
                                                     int M, int N, int K) {
  const int lane = threadIdx.x & 31;
  const int wave = threadIdx.x >> 5;
  const int lo   = lane & 15;
  const int hi   = lane >> 4;
  const int mBase = blockIdx.y * 64 + wave * 16;
  const int nBase = blockIdx.x * 64;

  const v8f zero8 = {0.f, 0.f, 0.f, 0.f, 0.f, 0.f, 0.f, 0.f};
  v8f acc[4];
#pragma unroll
  for (int t = 0; t < 4; ++t) acc[t] = zero8;

  const unsigned short* xr = X + (size_t)(mBase + lo) * K;
  for (int kk = 0; kk < K; kk += 32) {
    BF16x16 a;
    a.q[0] = *(const uint4*)(xr + kk + 8 * hi);
    a.q[1] = *(const uint4*)(xr + kk + 16 + 8 * hi);
#pragma unroll
    for (int t = 0; t < 4; ++t) {
      BF16x16 b;
      const unsigned short* wr = W + (size_t)(nBase + t * 16 + lo) * K + kk + 8 * hi;
      b.q[0] = *(const uint4*)(wr);
      b.q[1] = *(const uint4*)(wr + 16);
      acc[t] = WMMA_BF16(a.v, b.v, acc[t]);
    }
  }

#pragma unroll
  for (int t = 0; t < 4; ++t) {
#pragma unroll
    for (int r = 0; r < 8; ++r) {
      out[(size_t)(mBase + hi * 8 + r) * N + nBase + t * 16 + lo] = acc[t][r];
    }
  }
}

// ---------------------------------------------------------------------------
// Per-head RMSNorm + RoPE on Q/K, V converted + transposed to [HD][S].
// One thread = one (b, s, head) unit.
// ---------------------------------------------------------------------------
__global__ __launch_bounds__(256) void norm_rope_kernel(const float* __restrict__ QKV,
                                                        const float* __restrict__ qlnw,
                                                        const float* __restrict__ klnw,
                                                        unsigned short* __restrict__ Qr,
                                                        unsigned short* __restrict__ Kr,
                                                        unsigned short* __restrict__ VT) {
  const int NQ = BATCH * SEQ * NH;
  const int NK = BATCH * SEQ * NKV;
  int u = blockIdx.x * 256 + threadIdx.x;
  const float LOG1E4_D32 = 9.210340371976184f / 32.f;

  if (u < NQ) {
    int h = u % NH, s = (u / NH) % SEQ, b = u / (NH * SEQ);
    const float* x = QKV + (size_t)(b * SEQ + s) * NQKV + h * HD;
    float ss = 0.f;
    for (int j = 0; j < HD; ++j) { float v = x[j]; ss += v * v; }
    float rms = rsqrtf(ss * (1.0f / HD) + 1e-6f);
    unsigned short* o = Qr + ((size_t)(b * NH + h) * SEQ + s) * HD;
    for (int i = 0; i < 32; ++i) {
      float f = __expf(-(float)i * LOG1E4_D32);
      float c = cosf((float)s * f), sn = sinf((float)s * f);
      float a  = x[i] * rms * qlnw[i];
      float b2 = x[i + 32] * rms * qlnw[i + 32];
      o[i]      = f2bf(a * c - b2 * sn);
      o[i + 32] = f2bf(b2 * c + a * sn);
    }
  } else if (u < NQ + NK) {
    int uk = u - NQ;
    int kh = uk % NKV, s = (uk / NKV) % SEQ, b = uk / (NKV * SEQ);
    const float* x = QKV + (size_t)(b * SEQ + s) * NQKV + NH * HD + kh * HD;
    float ss = 0.f;
    for (int j = 0; j < HD; ++j) { float v = x[j]; ss += v * v; }
    float rms = rsqrtf(ss * (1.0f / HD) + 1e-6f);
    unsigned short* o = Kr + ((size_t)(b * NKV + kh) * SEQ + s) * HD;
    for (int i = 0; i < 32; ++i) {
      float f = __expf(-(float)i * LOG1E4_D32);
      float c = cosf((float)s * f), sn = sinf((float)s * f);
      float a  = x[i] * rms * klnw[i];
      float b2 = x[i + 32] * rms * klnw[i + 32];
      o[i]      = f2bf(a * c - b2 * sn);
      o[i + 32] = f2bf(b2 * c + a * sn);
    }
  } else if (u < NQ + 2 * NK) {
    int uv = u - NQ - NK;
    int kh = uv % NKV, s = (uv / NKV) % SEQ, b = uv / (NKV * SEQ);
    const float* x = QKV + (size_t)(b * SEQ + s) * NQKV + (NH + NKV) * HD + kh * HD;
    unsigned short* o = VT + (size_t)(b * NKV + kh) * HD * SEQ;
    for (int j = 0; j < HD; ++j) o[(size_t)j * SEQ + s] = f2bf(x[j]);
  }
}

// ---------------------------------------------------------------------------
// Sliding-window flash attention, one wave per (b, h, 16-query tile).
// S^T = K·Q^T  ->  queries on lanes, keys on VGPRs: softmax reductions are
// VGPR-local + one half-wave ds_swizzle.  O^T accumulated as V^T·P^T.
// No LDS, no divergence (EXEC all-ones for every WMMA).
// ---------------------------------------------------------------------------
__global__ __launch_bounds__(128) void attn_kernel(const unsigned short* __restrict__ Qr,
                                                   const unsigned short* __restrict__ Kr,
                                                   const unsigned short* __restrict__ VT,
                                                   unsigned short* __restrict__ Attn) {
  const int lane = threadIdx.x & 31;
  const int lo   = lane & 15;
  const int hi   = lane >> 4;
  const int gw   = blockIdx.x * 4 + (threadIdx.x >> 5);

  const int qt  = gw & (SEQ / 16 - 1);          // 128 query tiles
  const int h   = (gw >> 7) % NH;
  const int b   = gw / ((SEQ / 16) * NH);
  const int kvh = h / GROUPS;
  const int qb  = qt * 16;
  const int q   = qb + lo;

  const unsigned short* Qp = Qr + (size_t)(b * NH + h) * SEQ * HD;
  const unsigned short* Kp = Kr + (size_t)(b * NKV + kvh) * SEQ * HD;
  const unsigned short* Vp = VT + (size_t)(b * NKV + kvh) * HD * SEQ;

  // Q as WMMA B operand (lane = query col), both HD chunks (k=0..31, 32..63)
  BF16x16 qB0, qB1;
  {
    const unsigned short* qr = Qp + (size_t)q * HD;
    qB0.q[0] = *(const uint4*)(qr + 8 * hi);
    qB0.q[1] = *(const uint4*)(qr + 16 + 8 * hi);
    qB1.q[0] = *(const uint4*)(qr + 32 + 8 * hi);
    qB1.q[1] = *(const uint4*)(qr + 48 + 8 * hi);
  }

  const v8f zero8 = {0.f, 0.f, 0.f, 0.f, 0.f, 0.f, 0.f, 0.f};
  v8f accO[4];
#pragma unroll
  for (int t = 0; t < 4; ++t) accO[t] = zero8;
  float m = -3.0e38f, l = 0.f;

  int st = qb - (WIN - 1);
  if (st < 0) st = 0;
  st &= ~31;

  for (int kt = st; kt <= qb + 15; kt += 32) {
    // ---- S^T tiles: c0 = keys[kt..kt+15], c1 = keys[kt+16..kt+31]
    int k0r = kt + lo;      if (k0r > SEQ - 1) k0r = SEQ - 1;
    int k1r = kt + 16 + lo; if (k1r > SEQ - 1) k1r = SEQ - 1;
    const unsigned short* kr0 = Kp + (size_t)k0r * HD;
    const unsigned short* kr1 = Kp + (size_t)k1r * HD;
    BF16x16 kA0a, kA0b, kA1a, kA1b;
    kA0a.q[0] = *(const uint4*)(kr0 + 8 * hi);
    kA0a.q[1] = *(const uint4*)(kr0 + 16 + 8 * hi);
    kA0b.q[0] = *(const uint4*)(kr0 + 32 + 8 * hi);
    kA0b.q[1] = *(const uint4*)(kr0 + 48 + 8 * hi);
    kA1a.q[0] = *(const uint4*)(kr1 + 8 * hi);
    kA1a.q[1] = *(const uint4*)(kr1 + 16 + 8 * hi);
    kA1b.q[0] = *(const uint4*)(kr1 + 32 + 8 * hi);
    kA1b.q[1] = *(const uint4*)(kr1 + 48 + 8 * hi);

    v8f c0 = zero8, c1 = zero8;
    c0 = WMMA_BF16(kA0a.v, qB0.v, c0);
    c0 = WMMA_BF16(kA0b.v, qB1.v, c0);
    c1 = WMMA_BF16(kA1a.v, qB0.v, c1);
    c1 = WMMA_BF16(kA1b.v, qB1.v, c1);

    // ---- mask + scale; lane holds keys kt+hi*8+r (c0) and kt+16+hi*8+r (c1)
    float s0[8], s1[8];
    int a0f[8], a1f[8];
    float mloc = -3.0e38f;
#pragma unroll
    for (int r = 0; r < 8; ++r) {
      int key0 = kt + hi * 8 + r;
      int key1 = kt + 16 + hi * 8 + r;
      bool a0 = (key0 <= q) && (key0 > q - WIN);
      bool a1 = (key1 <= q) && (key1 > q - WIN);
      s0[r] = a0 ? c0[r] * 0.125f : -3.0e38f;
      s1[r] = a1 ? c1[r] * 0.125f : -3.0e38f;
      a0f[r] = a0; a1f[r] = a1;
      mloc = fmaxf(mloc, fmaxf(s0[r], s1[r]));
    }
    float mfull = fmaxf(mloc, swap16f(mloc));   // both key-halves of this query
    float mnew  = fmaxf(m, mfull);
    float alpha = __expf(m - mnew);

    float p0[8], p1[8];
    float sl = 0.f;
#pragma unroll
    for (int r = 0; r < 8; ++r) {
      p0[r] = a0f[r] ? __expf(s0[r] - mnew) : 0.f;
      p1[r] = a1f[r] ? __expf(s1[r] - mnew) : 0.f;
      sl += p0[r] + p1[r];
    }
    l = l * alpha + (sl + swap16f(sl));
    m = mnew;
#pragma unroll
    for (int t = 0; t < 4; ++t)
#pragma unroll
      for (int r = 0; r < 8; ++r) accO[t][r] *= alpha;

    // ---- build P^T as WMMA B operand (32 keys x 16 queries) via half-swap
    BF16x16 bp;
#pragma unroll
    for (int e = 0; e < 8; ++e) {
      float o0 = swap16f(p0[e]);                // keys 8..15 / 0..7 across halves
      float o1 = swap16f(p1[e]);                // keys 16..23 / 24..31
      float loS = hi ? o1 : p0[e];              // element e      -> key 16*hi+e
      float hiS = hi ? p1[e] : o0;              // element 8+e    -> key 16*hi+8+e
      bp.u[e]     = f2bf(loS);
      bp.u[8 + e] = f2bf(hiS);
    }

    // ---- O^T += V^T · P^T  (A: lane = head-dim row, contiguous keys)
#pragma unroll
    for (int t = 0; t < 4; ++t) {
      BF16x16 vA;
      const unsigned short* vr = Vp + (size_t)(t * 16 + lo) * SEQ + kt;
      vA.q[0] = *(const uint4*)(vr + 16 * hi);
      vA.q[1] = *(const uint4*)(vr + 16 * hi + 8);
      accO[t] = WMMA_BF16(vA.v, bp.v, accO[t]);
    }
  }

  // ---- normalize and write bf16 [B,S,NH*HD]; lane = query, VGPR = head dim
  float inv = 1.0f / l;
  unsigned short* orow = Attn + (size_t)(b * SEQ + q) * DIM + h * HD;
#pragma unroll
  for (int t = 0; t < 4; ++t) {
    unsigned int w0 = (unsigned)f2bf(accO[t][0] * inv) | ((unsigned)f2bf(accO[t][1] * inv) << 16);
    unsigned int w1 = (unsigned)f2bf(accO[t][2] * inv) | ((unsigned)f2bf(accO[t][3] * inv) << 16);
    unsigned int w2 = (unsigned)f2bf(accO[t][4] * inv) | ((unsigned)f2bf(accO[t][5] * inv) << 16);
    unsigned int w3 = (unsigned)f2bf(accO[t][6] * inv) | ((unsigned)f2bf(accO[t][7] * inv) << 16);
    *(uint4*)(orow + t * 16 + hi * 8) = make_uint4(w0, w1, w2, w3);
  }
}

// ---------------------------------------------------------------------------
// Host-side orchestration
// ---------------------------------------------------------------------------
extern "C" void kernel_launch(void* const* d_in, const int* in_sizes, int n_in,
                              void* d_out, int out_size, void* d_ws, size_t ws_size,
                              hipStream_t stream) {
  (void)in_sizes; (void)n_in; (void)out_size; (void)ws_size;
  const float* hs   = (const float*)d_in[0];
  const float* q_w  = (const float*)d_in[1];
  const float* k_w  = (const float*)d_in[2];
  const float* v_w  = (const float*)d_in[3];
  const float* o_w  = (const float*)d_in[4];
  const float* q_ln = (const float*)d_in[5];
  const float* k_ln = (const float*)d_in[6];
  float* out = (float*)d_out;

  char* ws = (char*)d_ws;
  unsigned short* Xbf  = (unsigned short*)(ws + 0);          //  4096x768  bf16
  unsigned short* Wqkv = (unsigned short*)(ws + 6291456);    //  1280x768  bf16
  unsigned short* Owt  = (unsigned short*)(ws + 8257536);    //   768x768  bf16
  float*          QKV  = (float*)        (ws + 9437184);     //  4096x1280 f32
  unsigned short* Qr   = (unsigned short*)(ws + 30408704);   //  B,NH,S,HD bf16
  unsigned short* Kr   = (unsigned short*)(ws + 36700160);   //  B,NKV,S,HD bf16
  unsigned short* VT   = (unsigned short*)(ws + 38797312);   //  B,NKV,HD,S bf16
  unsigned short* Attn = (unsigned short*)(ws + 40894464);   //  4096x768  bf16

  auto cvt = [&](const float* s, unsigned short* d, int n) {
    cvt_f32_bf16<<<(n + 255) / 256, 256, 0, stream>>>(s, d, n);
  };
  cvt(hs,  Xbf,               MROWS * DIM);          // hidden states
  cvt(q_w, Wqkv,              NH * HD * DIM);        // q_w rows 0..767
  cvt(k_w, Wqkv + 768 * DIM,  NKV * HD * DIM);       // k_w rows 768..1023
  cvt(v_w, Wqkv + 1024 * DIM, NKV * HD * DIM);       // v_w rows 1024..1279
  cvt(o_w, Owt,               DIM * DIM);

  // QKV = X @ Wcat^T  (M=4096, N=1280, K=768)
  gemm_xwt_bf16<<<dim3(NQKV / 64, MROWS / 64), 128, 0, stream>>>(Xbf, Wqkv, QKV,
                                                                 MROWS, NQKV, DIM);

  // RMSNorm + RoPE + V transpose: 49152 + 16384 + 16384 = 81920 units
  norm_rope_kernel<<<81920 / 256, 256, 0, stream>>>(QKV, q_ln, k_ln, Qr, Kr, VT);

  // 3072 waves, 4 per workgroup
  attn_kernel<<<(BATCH * NH * (SEQ / 16)) / 4, 128, 0, stream>>>(Qr, Kr, VT, Attn);

  // out = Attn @ o_w^T  (M=4096, N=768, K=768), f32 output
  gemm_xwt_bf16<<<dim3(DIM / 64, MROWS / 64), 128, 0, stream>>>(Attn, Owt, out,
                                                                MROWS, DIM, DIM);
}
